// SimpleCRAFeatures_55989193671249
// MI455X (gfx1250) — matrix operand order repaired and checked by reading, not compiled, not run on hardware
//
#include <hip/hip_runtime.h>
#include <hip/hip_bf16.h>

typedef __attribute__((ext_vector_type(16))) __bf16    v16bf;
typedef __attribute__((ext_vector_type(8)))  float     v8f;
typedef __attribute__((ext_vector_type(4)))  float     v4f;
typedef __attribute__((ext_vector_type(4)))  unsigned  v4u;
typedef __attribute__((ext_vector_type(8)))  int       v8i;
typedef __attribute__((ext_vector_type(4)))  int       v4i;

// float -> bf16 bits, round-to-nearest-even
__device__ __forceinline__ unsigned f2bf(float f) {
    unsigned u = __builtin_bit_cast(unsigned, f);
    unsigned r = u + 0x7FFFu + ((u >> 16) & 1u);
    return r >> 16;
}

#define FRAG_BYTES 131072           // 128 fragments x 1KB (bf16 codebook)
#define POOL_OFF   131072           // 8 waves x 16 words x 32 dims x 2B = 8KB
#define SMEM_BYTES 139264

// ---------------------------------------------------------------------------
// Kernel 1: per-codebook-row projection loss l_k = ||C_k Wp^T + b - C_k||^2
//           and codebook norms cn_k = ||C_k||^2.  128 blocks, one per code.
// ---------------------------------------------------------------------------
__global__ __launch_bounds__(256) void prep_proj_kernel(
    const float* __restrict__ cb, const float* __restrict__ pw,
    const float* __restrict__ pb, float* __restrict__ cn,
    float* __restrict__ lk) {
    __shared__ float s_sq[256];
    __shared__ float s_cn[256];
    const int k = blockIdx.x;     // 0..127
    const int t = threadIdx.x;    // 0..255
    float sq = 0.f, cnp = 0.f;
    const float* crow = cb + (size_t)k * 512;
    for (int d = t; d < 512; d += 256) {
        float p = pb[d];
        const float* wrow = pw + (size_t)d * 512;
        #pragma unroll 8
        for (int j = 0; j < 512; ++j) p = fmaf(crow[j], wrow[j], p);
        float c = crow[d];
        float e = p - c;
        sq  += e * e;
        cnp += c * c;
    }
    s_sq[t] = sq; s_cn[t] = cnp;
    __syncthreads();
    for (int s = 128; s > 0; s >>= 1) {
        if (t < s) { s_sq[t] += s_sq[t + s]; s_cn[t] += s_cn[t + s]; }
        __syncthreads();
    }
    if (t == 0) { lk[k] = s_sq[0]; cn[k] = s_cn[0]; }
}

// ---------------------------------------------------------------------------
// Kernel 2: pack codebook into bf16 WMMA B-fragments (fragment f = c*8 + t:
// K-chunk c of 32 dims, N-tile t of 16 codes).  B layout (32x16 bf16, wave32):
// lane L holds N = 16t+L%16; value j is K = 8(L/16)+j (j<8), 16+8(L/16)+(j-8) (j>=8).
// Stored per-lane contiguous: frags[f*512 + L*16 + j].
// ---------------------------------------------------------------------------
__global__ __launch_bounds__(256) void pack_cb_kernel(
    const float* __restrict__ cb, unsigned short* __restrict__ frags,
    float* __restrict__ loss_acc) {
    const int f  = blockIdx.x;        // 0..127
    const int c  = f >> 3;
    const int tl = f & 7;
    for (int e = threadIdx.x; e < 512; e += 256) {
        int L = e >> 4, j = e & 15;
        int code = 16 * tl + (L & 15);
        int dim  = 32 * c + ((j < 8) ? (8 * (L >> 4) + j) : (16 + 8 * (L >> 4) + (j - 8)));
        frags[(size_t)f * 512 + e] = (unsigned short)f2bf(cb[(size_t)code * 512 + dim]);
    }
    if (blockIdx.x == 0 && threadIdx.x == 0) *loss_acc = 0.f;
}

// ---------------------------------------------------------------------------
// Kernel 3: fused pool + distance WMMA + argmin + gather + loss partials.
// 256 blocks x 256 threads (8 waves); each wave owns 16 words (M=16),
// N = 128 codes (8 tiles), K = 512 (16 chunks of 32).
// 128KB fragment table staged once per block into LDS via the Tensor Data
// Mover; per-chunk pool staging and index exchange are wave-private (no
// block barriers inside the K loop).
// ---------------------------------------------------------------------------
__global__ __launch_bounds__(256) void cra_main_kernel(
    const float* __restrict__ chr,            // [8*8192*512] char embeddings
    const float* __restrict__ cb,             // fp32 codebook [128*512]
    const unsigned short* __restrict__ frags, // packed bf16 B fragments
    const float* __restrict__ cn,             // [128] codebook norms
    const float* __restrict__ lk,             // [128] per-code proj loss
    float* __restrict__ out_idx,              // [32768] indices (as float)
    float* __restrict__ out_emb,              // [32768*512]
    float* __restrict__ loss_acc) {
    extern __shared__ __align__(16) char smem[];
    unsigned short* s_frags = (unsigned short*)smem;                 // 128KB
    unsigned short (*pool)[16][32] =
        (unsigned short (*)[16][32])(smem + POOL_OFF);               // 8KB

    const int lane = threadIdx.x & 31;
    const int wv   = threadIdx.x >> 5;
    const int wordBase = blockIdx.x * 128 + wv * 16;

    // ---- stage the 128KB fragment table global -> LDS (once per block) ----
#if __has_builtin(__builtin_amdgcn_tensor_load_to_lds) && \
    __has_builtin(__builtin_amdgcn_s_wait_tensorcnt)
    if (wv == 0) {
        unsigned lds_base = (unsigned)(uintptr_t)(void*)s_frags;
        unsigned long long ga = (unsigned long long)(uintptr_t)frags;
        // D# group0: count=1, lds_addr[63:32], global_addr[120:64], type=2
        v4u g0 = { 1u, lds_base, (unsigned)(ga & 0xFFFFFFFFu),
                   (unsigned)((ga >> 32) & 0x01FFFFFFu) | (2u << 30) };
        // D# group1: data_size=3 (8B units); tensor_dim0 = tile_dim0 = 16384
        // (16384 * 8B = 128KB), tensor_dim1 = 1, dim0_stride = 16384.
        v8i g1 = { (int)(3u << 16),              // workgroup_mask=0, data_size=3
                   (int)(16384u << 16),          // tensor_dim0[15:0] @ bits63:48
                   (int)(1u << 16),              // dim0 hi=0 | tensor_dim1=1
                   (int)(16384u << 16),          // dim1 hi=0 | tile_dim0=16384
                   0,                            // tile_dim1=0, tile_dim2=0
                   16384,                        // tensor_dim0_stride lo
                   0, 0 };                       // stride hi / dim1_stride
        v4i gz4 = { 0, 0, 0, 0 };                // groups 2/3 unused (<=2D)
        v8i gz8 = { 0, 0, 0, 0, 0, 0, 0, 0 };    // trailing group (unused)
        __builtin_amdgcn_tensor_load_to_lds(g0, g1, gz4, gz4, gz8, 0);
        __builtin_amdgcn_s_wait_tensorcnt(0);
    }
#else
    for (int i = threadIdx.x; i < FRAG_BYTES / 16; i += 256)
        ((uint4*)s_frags)[i] = ((const uint4*)frags)[i];
#endif
    __syncthreads();   // fragment table is shared by all 8 waves

    v8f acc[8];
    #pragma unroll
    for (int t = 0; t < 8; ++t) acc[t] = (v8f){0.f,0.f,0.f,0.f,0.f,0.f,0.f,0.f};

    const int m16   = lane & 15;
    const int abase = (lane < 16) ? 0 : 16;   // A-frag byte base (ISA 7.12.2)

    for (int c = 0; c < 16; ++c) {
        // --- stage: pool adjacent char pairs (float4 loads), bf16 to LDS ---
        #pragma unroll
        for (int j = 0; j < 4; ++j) {
            int i = lane + 32 * j;            // 128 float4 per wave-chunk
            int m = i >> 3, q = i & 7;
            const v4f* pa = (const v4f*)chr +
                            (size_t)(wordBase + m) * 256 + c * 8 + q;
            v4f a = pa[0];
            v4f b = pa[128];                  // second char of the pair
            if (c < 12) {                     // prefetch 4 chunks (512B) ahead
                __builtin_prefetch((const void*)(pa + 32), 0, 0);
                __builtin_prefetch((const void*)(pa + 160), 0, 0);
            }
            unsigned lo = f2bf(0.5f * (a.x + b.x)) | (f2bf(0.5f * (a.y + b.y)) << 16);
            unsigned hi = f2bf(0.5f * (a.z + b.z)) | (f2bf(0.5f * (a.w + b.w)) << 16);
            uint2 pk; pk.x = lo; pk.y = hi;
            *(uint2*)&pool[wv][m][q * 4] = pk;    // ds_store_b64
        }
        __builtin_amdgcn_wave_barrier();  // wave-private LDS: order store->load

        // --- A fragment: two contiguous 16B LDS reads per lane ---
        union { v16bf v; uint4 q[2]; } af;
        const char* prow = (const char*)&pool[wv][m16][0];
        af.q[0] = *(const uint4*)(prow + abase);        // ds_load_b128
        af.q[1] = *(const uint4*)(prow + 32 + abase);   // ds_load_b128

        // --- 8 N-tiles of v_wmma_f32_16x16x32_bf16, B straight from LDS ---
        #pragma unroll
        for (int t = 0; t < 8; ++t) {
            union { v16bf v; uint4 q[2]; } bf;
            const uint4* src =
                (const uint4*)(s_frags + (size_t)(c * 8 + t) * 512 + lane * 16);
            bf.q[0] = src[0];
            bf.q[1] = src[1];
            acc[t] = __builtin_amdgcn_wmma_f32_16x16x32_bf16(
                false, af.v, false, bf.v, (short)0, acc[t], false, false);
        }
        __builtin_amdgcn_wave_barrier();  // keep next-chunk stores below loads
    }

    // --- argmin_k ( ||c_k||^2 - 2 w.c_k )  (||w||^2 dropped: constant) ---
    // C/D layout: lane (mod 16) = N, VGPR r = M row (r / 8+r per half-wave).
    float cnv[8];
    #pragma unroll
    for (int t = 0; t < 8; ++t) cnv[t] = cn[16 * t + m16];

    int   rowidx[8];
    float lsum = 0.f;
    #pragma unroll
    for (int r = 0; r < 8; ++r) {
        float bestv = 3.4e38f;
        int   besti = 0;
        #pragma unroll
        for (int t = 0; t < 8; ++t) {
            float v = fmaf(-2.0f, acc[t][r], cnv[t]);
            int   n = 16 * t + m16;
            if (v < bestv) { bestv = v; besti = n; }
        }
        // butterfly min within each 16-lane half (rows r and 8+r)
        #pragma unroll
        for (int off = 1; off < 16; off <<= 1) {
            float ov = __shfl_xor(bestv, off, 16);
            int   oi = __shfl_xor(besti, off, 16);
            if (ov < bestv || (ov == bestv && oi < besti)) { bestv = ov; besti = oi; }
        }
        rowidx[r] = besti;   // all lanes of each half hold their row's winner
        int row = (lane < 16) ? r : (8 + r);
        if (m16 == r) {
            out_idx[wordBase + row] = (float)besti;
            lsum += lk[besti];
        }
    }

    // one atomic per wave for the loss partial
    #pragma unroll
    for (int off = 1; off < 32; off <<= 1) lsum += __shfl_xor(lsum, off, 32);
    if (lane == 0) atomicAdd(loss_acc, lsum);

    // --- gather fp32 codebook rows -> word_embeddings (NT v4f stores) ---
    #pragma unroll 2
    for (int m = 0; m < 16; ++m) {
        int code = __shfl(rowidx[m & 7], (m < 8) ? 0 : 16, 32);
        const v4f* src = (const v4f*)(cb + (size_t)code * 512);
        v4f* dst = (v4f*)(out_emb + (size_t)(wordBase + m) * 512);
        #pragma unroll
        for (int i = lane; i < 128; i += 32)
            __builtin_nontemporal_store(src[i], dst + i);
    }
}

// ---------------------------------------------------------------------------
// Kernel 4: finalize scalar loss = sum_w l[idx_w] / (BW * D)
// ---------------------------------------------------------------------------
__global__ void finalize_kernel(const float* __restrict__ loss_acc,
                                float* __restrict__ out_loss) {
    if (threadIdx.x == 0 && blockIdx.x == 0)
        out_loss[0] = loss_acc[0] * (1.0f / (32768.0f * 512.0f));
}

// ---------------------------------------------------------------------------
extern "C" void kernel_launch(void* const* d_in, const int* in_sizes, int n_in,
                              void* d_out, int out_size, void* d_ws, size_t ws_size,
                              hipStream_t stream) {
    // inputs in setup_inputs() order:
    // 0: char_tokens (int, unused by the reference math)
    // 1: char_embeddings f32 [8,8192,512]
    // 2: word_codebook   f32 [128,512]
    // 3: proj_w          f32 [512,512]
    // 4: proj_b          f32 [512]
    const float* chr = (const float*)d_in[1];
    const float* cb  = (const float*)d_in[2];
    const float* pw  = (const float*)d_in[3];
    const float* pb  = (const float*)d_in[4];

    float* out      = (float*)d_out;
    float* out_idx  = out;                       // 32768
    float* out_emb  = out + 32768;               // 32768*512
    float* out_loss = out + 32768 + 32768 * 512; // 1

    char* ws = (char*)d_ws;
    unsigned short* frags = (unsigned short*)(ws);          // 131072 B
    float* cnv      = (float*)(ws + 131072);                // 512 B
    float* lkv      = (float*)(ws + 131584);                // 512 B
    float* loss_acc = (float*)(ws + 132096);                // 4 B

    prep_proj_kernel<<<128, 256, 0, stream>>>(cb, pw, pb, cnv, lkv);
    pack_cb_kernel<<<128, 256, 0, stream>>>(cb, frags, loss_acc);
    cra_main_kernel<<<256, 256, SMEM_BYTES, stream>>>(chr, cb, frags, cnv, lkv,
                                                      out_idx, out_emb, loss_acc);
    finalize_kernel<<<1, 32, 0, stream>>>(loss_acc, out_loss);
}
